// MultiHeadAttention_56341380989271
// MI455X (gfx1250) — compile-verified, matrix operand order checked by reading, not run on hardware
//
#include <hip/hip_runtime.h>

// Problem constants (from reference)
#define B_   2
#define L_   2048
#define HID_ 1024
#define NH_  16
#define HD_  64
#define ML_  (B_ * L_)      // 4096 rows in the flattened GEMMs

typedef __attribute__((ext_vector_type(16))) __bf16 v16bf;
typedef __attribute__((ext_vector_type(8)))  __bf16 v8bf;
typedef __attribute__((ext_vector_type(8)))  float  v8f;

union Frag { v16bf v; v8bf h[2]; };

// ---------------------------------------------------------------------------
// 1) fp32 -> bf16 convert
// ---------------------------------------------------------------------------
__global__ void cvt_f32_bf16(const float* __restrict__ src,
                             unsigned short* __restrict__ dst, int n) {
    int i = blockIdx.x * blockDim.x + threadIdx.x;
    if (i < n) {
        ((__bf16*)dst)[i] = (__bf16)src[i];
    }
}

// ---------------------------------------------------------------------------
// Async global->LDS staging of a 128-row x 32-col bf16 tile (8 KB).
// 128 threads, each issues 4 GLOBAL_LOAD_ASYNC_TO_LDS_B128 (ASYNCcnt-tracked).
// ---------------------------------------------------------------------------
__device__ __forceinline__ void stage_tile_async(const __bf16* __restrict__ g,
                                                 int ldK, int k0,
                                                 __bf16* lds_base, int t) {
#pragma unroll
    for (int r = 0; r < 4; ++r) {
        const int e     = r * 128 + t;      // [0,512): 128 rows x 4 16B chunks
        const int row   = e >> 2;
        const int chunk = e & 3;
        const __bf16* gp = g + (size_t)row * ldK + k0 + chunk * 8;
        const unsigned int lds =
            (unsigned int)(uintptr_t)(lds_base + row * 32 + chunk * 8);
        asm volatile("global_load_async_to_lds_b128 %0, %1, off"
                     :: "v"(lds), "v"(gp) : "memory");
    }
}

// ---------------------------------------------------------------------------
// 2) bf16 WMMA GEMM:  C[M,N] = A[M,K] @ W[N,K]^T + bias[N]
//    Block = 4 waves (2x2) -> 128x128 tile; A/W staged through double-buffered
//    LDS with async loads; each wave runs a 64x64 / 16-accumulator WMMA tile.
//    mode 0: bf16 out -> [B,NH,L,HD]   (Q, K projections)
//    mode 1: bf16 out -> [B,NH,HD,L]   (V projection, transposed)
//    mode 2: fp32 out -> row-major [M,N] (final output projection)
// ---------------------------------------------------------------------------
__global__ __launch_bounds__(128, 1)
void gemm64_bf16(const unsigned short* __restrict__ Ap,
                 const unsigned short* __restrict__ Wp,
                 const float* __restrict__ bias,
                 void* __restrict__ outp,
                 int M, int N, int K, int mode) {
    __shared__ __align__(16) __bf16 As[2][128 * 32];   // 2 x 8 KB
    __shared__ __align__(16) __bf16 Ws[2][128 * 32];   // 2 x 8 KB

    const __bf16* A = (const __bf16*)Ap;
    const __bf16* W = (const __bf16*)Wp;

    const int t    = threadIdx.x;
    const int wid  = t >> 5;            // wave in block [0,4)
    const int lane = t & 31;
    const int hi16 = lane >> 4;
    const int lr   = lane & 15;
    const int wm   = wid >> 1;          // wave row in 2x2
    const int wn   = wid & 1;           // wave col in 2x2

    const int nb = N >> 7;              // 128-wide N blocks
    const int Mb = (blockIdx.x / nb) * 128;
    const int Nb = (blockIdx.x % nb) * 128;
    const int m0 = Mb + wm * 64;
    const int n0 = Nb + wn * 64;

    const __bf16* Ablk = A + (size_t)Mb * K;
    const __bf16* Wblk = W + (size_t)Nb * K;

    v8f c[4][4] = {};

    // Prologue: stage K-step 0 into buffer 0
    stage_tile_async(Ablk, K, 0, &As[0][0], t);
    stage_tile_async(Wblk, K, 0, &Ws[0][0], t);

    int buf = 0;
    for (int k0 = 0; k0 < K; k0 += 32, buf ^= 1) {
        // my async writes into `buf` are done; barrier also separates all
        // waves' previous-iteration DS reads from the staging below
        asm volatile("s_wait_asynccnt 0" ::: "memory");
        __syncthreads();

        if (k0 + 32 < K) {
            stage_tile_async(Ablk, K, k0 + 32, &As[buf ^ 1][0], t);
            stage_tile_async(Wblk, K, k0 + 32, &Ws[buf ^ 1][0], t);
        }

        const __bf16* as = &As[buf][0];
        const __bf16* ws = &Ws[buf][0];
        Frag a[4], b[4];
#pragma unroll
        for (int i = 0; i < 4; ++i) {
            const __bf16* pa = as + (wm * 64 + 16 * i + lr) * 32 + hi16 * 8;
            a[i].h[0] = *(const v8bf*)pa;
            a[i].h[1] = *(const v8bf*)(pa + 16);
            const __bf16* pb = ws + (wn * 64 + 16 * i + lr) * 32 + hi16 * 8;
            b[i].h[0] = *(const v8bf*)pb;
            b[i].h[1] = *(const v8bf*)(pb + 16);
        }
#pragma unroll
        for (int i = 0; i < 4; ++i)
#pragma unroll
            for (int j = 0; j < 4; ++j)
                c[i][j] = __builtin_amdgcn_wmma_f32_16x16x32_bf16(
                    false, a[i].v, false, b[j].v, (short)0, c[i][j], false, false);
    }

    // Epilogue: C element r of v8f <-> row r + 8*hi16, col = lr (ISA 7.12.2)
#pragma unroll
    for (int i = 0; i < 4; ++i) {
#pragma unroll
        for (int j = 0; j < 4; ++j) {
            const int ncol = n0 + 16 * j + lr;
            const float bv = bias[ncol];
#pragma unroll
            for (int r = 0; r < 8; ++r) {
                const int m  = m0 + 16 * i + r + 8 * hi16;
                const float val = c[i][j][r] + bv;
                if (mode == 2) {
                    ((float*)outp)[(size_t)m * N + ncol] = val;
                } else {
                    const int bb = m >> 11;          // m / L
                    const int ll = m & (L_ - 1);
                    const int hh = ncol >> 6;        // n / HD
                    const int dd = ncol & (HD_ - 1);
                    size_t idx;
                    if (mode == 0)
                        idx = (((size_t)(bb * NH_ + hh)) * L_ + ll) * HD_ + dd;
                    else
                        idx = (((size_t)(bb * NH_ + hh)) * HD_ + dd) * L_ + ll;
                    ((__bf16*)outp)[idx] = (__bf16)val;
                }
            }
        }
    }
}

// ---------------------------------------------------------------------------
// 3) Flash attention: one wave per (b, h, 16-row query tile).
//    Q,K in [B,NH,L,HD]; V transposed [B,NH,HD,L]; online softmax in f32.
// ---------------------------------------------------------------------------
__global__ __launch_bounds__(128, 1)
void attn_flash(const unsigned short* __restrict__ Qp,
                const unsigned short* __restrict__ Kp,
                const unsigned short* __restrict__ Vtp,
                const unsigned char* __restrict__ kmask,
                unsigned short* __restrict__ attp) {
    __shared__ __align__(16) __bf16 plds[4][16 * 32];   // 1 KB per wave

    const __bf16* Q  = (const __bf16*)Qp;
    const __bf16* Kb = (const __bf16*)Kp;
    const __bf16* Vt = (const __bf16*)Vtp;
    __bf16* att      = (__bf16*)attp;

    const int wave = blockIdx.x * (blockDim.x >> 5) + (threadIdx.x >> 5);
    const int wl   = threadIdx.x >> 5;
    const int lane = threadIdx.x & 31;
    const int hi16 = lane >> 4;
    const int lr   = lane & 15;

    const int qt = wave & (L_ / 16 - 1);      // query tile   [0,128)
    const int bh = wave >> 7;                 // (b*NH + h)   [0,32)
    const int h  = bh & (NH_ - 1);
    const int b  = bh >> 4;
    const int q0 = qt * 16;

    const __bf16* Qbh = Q  + (size_t)bh * L_ * HD_;
    const __bf16* Kbh = Kb + (size_t)bh * L_ * HD_;
    const __bf16* Vbh = Vt + (size_t)bh * HD_ * L_;     // [HD][L]

    // Q tile: 2 A-fragments (16 x 32 each over head dim)
    Frag qa[2];
#pragma unroll
    for (int cch = 0; cch < 2; ++cch) {
        const __bf16* p = Qbh + (size_t)(q0 + lr) * HD_ + 32 * cch + hi16 * 8;
        qa[cch].h[0] = *(const v8bf*)p;
        qa[cch].h[1] = *(const v8bf*)(p + 16);
    }

    v8f acc[4] = {};
    float mrow[8], lrow[8];
#pragma unroll
    for (int j = 0; j < 8; ++j) { mrow[j] = -1e30f; lrow[j] = 0.0f; }
    const float scale = 0.125f;               // 1/sqrt(64)

    const int kvend = q0 + 16;                // causal bound
    for (int kv0 = 0; kv0 < kvend; kv0 += 32) {
        // ---- scores: 2 tiles of 16q x 16kv, each accumulated over 2 hd-chunks
        v8f s[2];
#pragma unroll
        for (int tt = 0; tt < 2; ++tt) {
            Frag k0f, k1f;
            const __bf16* pk = Kbh + (size_t)(kv0 + 16 * tt + lr) * HD_ + hi16 * 8;
            k0f.h[0] = *(const v8bf*)pk;        k0f.h[1] = *(const v8bf*)(pk + 16);
            k1f.h[0] = *(const v8bf*)(pk + 32); k1f.h[1] = *(const v8bf*)(pk + 48);
            v8f z = {};
            z = __builtin_amdgcn_wmma_f32_16x16x32_bf16(
                    false, qa[0].v, false, k0f.v, (short)0, z, false, false);
            z = __builtin_amdgcn_wmma_f32_16x16x32_bf16(
                    false, qa[1].v, false, k1f.v, (short)0, z, false, false);
            s[tt] = z;
        }

        // ---- mask + scale
        int  kvi[2]; bool km[2];
#pragma unroll
        for (int tt = 0; tt < 2; ++tt) {
            kvi[tt] = kv0 + 16 * tt + lr;
            km[tt]  = kmask[b * L_ + kvi[tt]] != 0;
        }
        float p[2][8], mloc[8];
#pragma unroll
        for (int j = 0; j < 8; ++j) {
            const int qi = q0 + j + 8 * hi16;
            float best = -1e30f;
#pragma unroll
            for (int tt = 0; tt < 2; ++tt) {
                const bool valid = km[tt] && (kvi[tt] <= qi);
                const float sv = valid ? s[tt][j] * scale : -1e30f;
                p[tt][j] = sv;
                best = fmaxf(best, sv);
            }
            mloc[j] = best;
        }
        // row-max across the 16 lanes sharing a row group (wave32 halves)
#pragma unroll
        for (int off = 1; off < 16; off <<= 1)
#pragma unroll
            for (int j = 0; j < 8; ++j)
                mloc[j] = fmaxf(mloc[j], __shfl_xor(mloc[j], off, 32));

        float alpha[8];
#pragma unroll
        for (int j = 0; j < 8; ++j) {
            const float mn = fmaxf(mrow[j], mloc[j]);
            alpha[j] = __expf(mrow[j] - mn);
            mrow[j]  = mn;
        }
        float rsum[8];
#pragma unroll
        for (int j = 0; j < 8; ++j) {
            float r = 0.0f;
#pragma unroll
            for (int tt = 0; tt < 2; ++tt) {
                const float e = (p[tt][j] > -1e29f) ? __expf(p[tt][j] - mrow[j]) : 0.0f;
                p[tt][j] = e;
                r += e;
            }
            rsum[j] = r;
        }
#pragma unroll
        for (int off = 1; off < 16; off <<= 1)
#pragma unroll
            for (int j = 0; j < 8; ++j)
                rsum[j] += __shfl_xor(rsum[j], off, 32);
#pragma unroll
        for (int j = 0; j < 8; ++j) lrow[j] = lrow[j] * alpha[j] + rsum[j];
#pragma unroll
        for (int i = 0; i < 4; ++i)
#pragma unroll
            for (int j = 0; j < 8; ++j) acc[i][j] = acc[i][j] * alpha[j];

        // ---- transpose P (C-layout -> A-layout) through per-wave LDS slab
        __bf16* pl = &plds[wl][0];
#pragma unroll
        for (int tt = 0; tt < 2; ++tt)
#pragma unroll
            for (int j = 0; j < 8; ++j)
                pl[(j + 8 * hi16) * 32 + 16 * tt + lr] = (__bf16)p[tt][j];
        asm volatile("s_wait_dscnt 0" ::: "memory");

        Frag pa;
        {
            const __bf16* pp = pl + (size_t)lr * 32 + hi16 * 8;
            pa.h[0] = *(const v8bf*)pp;
            pa.h[1] = *(const v8bf*)(pp + 16);
        }
        asm volatile("s_wait_dscnt 0" ::: "memory");

        // ---- out += P (16x32 over kv) @ V (32kv x 16hd), 4 hd tiles
#pragma unroll
        for (int i = 0; i < 4; ++i) {
            Frag vb;
            const __bf16* pv = Vbh + (size_t)(16 * i + lr) * L_ + kv0 + hi16 * 8;
            vb.h[0] = *(const v8bf*)pv;
            vb.h[1] = *(const v8bf*)(pv + 16);
            acc[i] = __builtin_amdgcn_wmma_f32_16x16x32_bf16(
                false, pa.v, false, vb.v, (short)0, acc[i], false, false);
        }
    }

    // ---- finalize: divide by max(rowsum, 1e-6), write [B, L, HID] bf16
#pragma unroll
    for (int j = 0; j < 8; ++j) lrow[j] = 1.0f / fmaxf(lrow[j], 1e-6f);
#pragma unroll
    for (int i = 0; i < 4; ++i)
#pragma unroll
        for (int j = 0; j < 8; ++j) {
            const float v = acc[i][j] * lrow[j];
            const int qi  = q0 + j + 8 * hi16;
            const int dd  = 16 * i + lr;
            att[((size_t)b * L_ + qi) * HID_ + h * HD_ + dd] = (__bf16)v;
        }
}

// ---------------------------------------------------------------------------
// Launch
// ---------------------------------------------------------------------------
extern "C" void kernel_launch(void* const* d_in, const int* in_sizes, int n_in,
                              void* d_out, int out_size, void* d_ws, size_t ws_size,
                              hipStream_t stream) {
    const float* q   = (const float*)d_in[0];
    const float* k   = (const float*)d_in[1];
    const float* v   = (const float*)d_in[2];
    const unsigned char* key_mask = (const unsigned char*)d_in[3];
    // d_in[4] = attn_mask (causal triu) -- computed from indices instead
    const float* Wq = (const float*)d_in[5];
    const float* bq = (const float*)d_in[6];
    const float* Wk = (const float*)d_in[7];
    const float* bk = (const float*)d_in[8];
    const float* Wv = (const float*)d_in[9];
    const float* bv = (const float*)d_in[10];
    const float* Wo = (const float*)d_in[11];
    const float* bo = (const float*)d_in[12];

    const size_t nX = (size_t)B_ * L_ * HID_;   // 4,194,304
    const size_t nW = (size_t)HID_ * HID_;      // 1,048,576

    unsigned short* qb  = (unsigned short*)d_ws;
    unsigned short* kb  = qb  + nX;
    unsigned short* vb  = kb  + nX;
    unsigned short* wqb = vb  + nX;
    unsigned short* wkb = wqb + nW;
    unsigned short* wvb = wkb + nW;
    unsigned short* wob = wvb + nW;
    unsigned short* Qp  = wob + nW;
    unsigned short* Kp  = Qp  + nX;
    unsigned short* Vt  = Kp  + nX;
    unsigned short* att = Vt  + nX;

    // 1) fp32 -> bf16 conversions
    cvt_f32_bf16<<<(int)(nX / 256), 256, 0, stream>>>(q,  qb,  (int)nX);
    cvt_f32_bf16<<<(int)(nX / 256), 256, 0, stream>>>(k,  kb,  (int)nX);
    cvt_f32_bf16<<<(int)(nX / 256), 256, 0, stream>>>(v,  vb,  (int)nX);
    cvt_f32_bf16<<<(int)(nW / 256), 256, 0, stream>>>(Wq, wqb, (int)nW);
    cvt_f32_bf16<<<(int)(nW / 256), 256, 0, stream>>>(Wk, wkb, (int)nW);
    cvt_f32_bf16<<<(int)(nW / 256), 256, 0, stream>>>(Wv, wvb, (int)nW);
    cvt_f32_bf16<<<(int)(nW / 256), 256, 0, stream>>>(Wo, wob, (int)nW);

    // 2) projections: 128x128 block tiles -> (4096/128)*(1024/128) = 256 blocks
    const int gemmBlocks = (ML_ / 128) * (HID_ / 128);   // 256
    gemm64_bf16<<<gemmBlocks, 128, 0, stream>>>(qb, wqb, bq, Qp, ML_, HID_, HID_, 0);
    gemm64_bf16<<<gemmBlocks, 128, 0, stream>>>(kb, wkb, bk, Kp, ML_, HID_, HID_, 0);
    gemm64_bf16<<<gemmBlocks, 128, 0, stream>>>(vb, wvb, bv, Vt, ML_, HID_, HID_, 1);

    // 3) attention: B*NH*(L/16) = 4096 waves = 1024 blocks x 4 waves
    attn_flash<<<B_ * NH_ * (L_ / 16) / 4, 128, 0, stream>>>(Qp, Kp, Vt, key_mask, att);

    // 4) output projection -> fp32 d_out
    gemm64_bf16<<<gemmBlocks, 128, 0, stream>>>(att, wob, bo, (float*)d_out,
                                                ML_, HID_, HID_, 2);
}
